// SimpleSelfAttention_87995289960765
// MI455X (gfx1250) — matrix operand order repaired and checked by reading, not compiled, not run on hardware
//
#include <hip/hip_runtime.h>

typedef __bf16 bf16_t;
typedef unsigned short u16;
typedef unsigned int   u32;
typedef __attribute__((ext_vector_type(16))) __bf16 v16bf;
typedef __attribute__((ext_vector_type(8)))  __bf16 v8bf;
typedef __attribute__((ext_vector_type(8)))  float  v8f;

#define WMMA_BF16(A, B, C) \
  __builtin_amdgcn_wmma_f32_16x16x32_bf16(false, (A), false, (B), (short)0, (C), false, false)

static constexpr int S_LEN = 2048;   // sequence length
static constexpr int D_DIM = 128;    // embed dim
static constexpr int TQ    = 128;    // Q rows per workgroup (32 per wave x 4 waves)
static constexpr int TT    = 32;     // K/V rows per iteration
static constexpr int KSTR  = 136;    // padded LDS stride (bf16 elems) for K rows (272B, 16B-aligned, bank-staggered)
static constexpr int VSTRU = 36;     // padded LDS stride (u32 elems) for V^T rows (144B, bank-staggered)
static constexpr int PSTRU = 36;     // padded LDS stride (u32 elems) for P staging rows

__device__ __forceinline__ v16bf cat8(v8bf a, v8bf b) {
  return __builtin_shufflevector(a, b, 0,1,2,3,4,5,6,7,8,9,10,11,12,13,14,15);
}
__device__ __forceinline__ v16bf ld16(const bf16_t* p) {
  v8bf a = *reinterpret_cast<const v8bf*>(p);
  v8bf b = *reinterpret_cast<const v8bf*>(p + 8);
  return cat8(a, b);
}
// 8 interleaved {hi,lo} u32 pairs -> separated v8bf hi / v8bf lo
__device__ __forceinline__ void unpack8(const u32* p, v8bf& hi, v8bf& lo) {
  v8bf q0 = __builtin_bit_cast(v8bf, *reinterpret_cast<const uint4*>(p));
  v8bf q1 = __builtin_bit_cast(v8bf, *reinterpret_cast<const uint4*>(p + 4));
  hi = __builtin_shufflevector(q0, q1, 0,2,4,6,8,10,12,14);
  lo = __builtin_shufflevector(q0, q1, 1,3,5,7,9,11,13,15);
}
__device__ __forceinline__ void unpack16(const u32* p, v16bf& hi, v16bf& lo) {
  v8bf h0, l0, h1, l1;
  unpack8(p, h0, l0);
  unpack8(p + 8, h1, l1);
  hi = cat8(h0, h1);
  lo = cat8(l0, l1);
}

// ---- Pre-pass: global logit bound. max over rows of ||x||^2 -------------------
__global__ void zero_ws_kernel(u32* ws) { ws[0] = 0u; }

__global__ __launch_bounds__(128)
void maxnorm2_kernel(const float* __restrict__ in, u32* __restrict__ ws) {
  const int w    = threadIdx.x >> 5;
  const int lane = threadIdx.x & 31;
  const int waveId = blockIdx.x * 4 + w;           // 0..1023
  float mx = 0.0f;
  for (int i = 0; i < 32; ++i) {
    const size_t row = (size_t)waveId * 32 + i;    // 0..32767
    const float4 v = *reinterpret_cast<const float4*>(in + row * D_DIM + lane * 4);
    float s = v.x * v.x + v.y * v.y + v.z * v.z + v.w * v.w;
    s += __shfl_xor(s, 1, 32);
    s += __shfl_xor(s, 2, 32);
    s += __shfl_xor(s, 4, 32);
    s += __shfl_xor(s, 8, 32);
    s += __shfl_xor(s, 16, 32);
    mx = fmaxf(mx, s);
  }
  if (lane == 0) atomicMax(ws, __float_as_uint(mx));  // positive floats: bit order == value order
}

// ---- Main fused attention kernel ---------------------------------------------
__global__ __launch_bounds__(128)
void flash_attn_bf16x3_kernel(const float* __restrict__ in, float* __restrict__ out,
                              const u32* __restrict__ ws) {
  // K row-major hi/lo planes; V transposed + P staging with hi/lo interleaved per u32
  __shared__ __align__(16) bf16_t sKhi[TT * KSTR];
  __shared__ __align__(16) bf16_t sKlo[TT * KSTR];
  __shared__ __align__(16) u32    sVt[D_DIM * VSTRU];
  __shared__ __align__(16) u32    sP[4][32 * PSTRU];

  const int tid  = threadIdx.x;
  const int w    = tid >> 5;          // wave id, 0..3
  const int lane = tid & 31;
  const int m    = lane & 15;         // row-in-tile / B column / N column
  const int hh   = lane >> 4;         // lane half (K-half selector in WMMA layouts)
  const int b    = blockIdx.y;        // batch
  const int row0 = blockIdx.x * TQ + w * 32;   // 32 Q rows per wave (two 16-row A tiles)

  // logits in log2 units: scale = (1/sqrt(128)) * log2(e); softmax invariant
  const float scale = 0.088388347648318447f * 1.4426950408889634f;
  // Static per-dispatch logit upper bound: q.k <= max||x||^2. Subtracting a
  // constant is an exact power-of-2 scaling -> softmax ratios unchanged; the
  // bound guarantees exp2 argument <= 0 and gap << 126, so no over/underflow.
  const float negM = -(__uint_as_float(ws[0]) * scale);

  const float* batchBase = in + (size_t)b * S_LEN * D_DIM;

  // ---- Preload two Q A-tiles as bf16 hi/lo -----------------------------------
  // A operand k order used here (contiguous, not the ISA table order): lane half hh: elem j -> K = hh*16+j
  v16bf aQhi[2][4], aQlo[2][4];
  #pragma unroll
  for (int g = 0; g < 2; ++g) {
    const float* qrow = batchBase + (size_t)(row0 + g * 16 + m) * D_DIM;
    #pragma unroll
    for (int dc = 0; dc < 4; ++dc) {
      #pragma unroll
      for (int j = 0; j < 16; ++j) {
        const int k = dc * 32 + hh * 16 + j;
        const float x = qrow[k];
        const bf16_t hi = (bf16_t)x;
        aQhi[g][dc][j] = hi;
        aQlo[g][dc][j] = (bf16_t)(x - (float)hi);
      }
    }
  }

  // All-ones B operand: every column of D = row sum of A  ->  rowSum via WMMA
  v16bf onesB;
  #pragma unroll
  for (int j = 0; j < 16; ++j) onesB[j] = (bf16_t)1.0f;

  // ---- Accumulators: O tiles and row-sum tiles --------------------------------
  v8f o[2][8], sumAcc[2];
  #pragma unroll
  for (int g = 0; g < 2; ++g) {
    sumAcc[g] = (v8f){0,0,0,0,0,0,0,0};
    #pragma unroll
    for (int dt = 0; dt < 8; ++dt) o[g][dt] = (v8f){0,0,0,0,0,0,0,0};
  }

  const int srow  = tid >> 2;   // 0..31 : staging row
  const int squad = tid & 3;    // 0..3  : staging quad group

  // ---- Software pipeline: tile t held in registers, staged at top of iter ----
  float4 reg[8];
  #pragma unroll
  for (int i = 0; i < 8; ++i)
    reg[i] = *reinterpret_cast<const float4*>(batchBase + (size_t)srow * D_DIM + (squad + 4 * i) * 4);

  for (int t0 = 0; t0 < S_LEN; t0 += TT) {
    __syncthreads();   // sync1: all LDS readers of previous tile are done

    // ---- Stage tile t (registers -> bf16 hi/lo in LDS) -------------------------
    #pragma unroll
    for (int i = 0; i < 8; ++i) {
      const int d0 = (squad + 4 * i) * 4;
      const float xs[4] = {reg[i].x, reg[i].y, reg[i].z, reg[i].w};
      u16 kh[4], kl[4];
      #pragma unroll
      for (int e = 0; e < 4; ++e) {
        const float  x  = xs[e];
        const bf16_t hi = (bf16_t)x;
        const bf16_t lo = (bf16_t)(x - (float)hi);
        kh[e] = __builtin_bit_cast(u16, hi);
        kl[e] = __builtin_bit_cast(u16, lo);
        sVt[(d0 + e) * VSTRU + srow] = (u32)kh[e] | ((u32)kl[e] << 16);
      }
      *reinterpret_cast<ushort4*>(&sKhi[srow * KSTR + d0]) = make_ushort4(kh[0], kh[1], kh[2], kh[3]);
      *reinterpret_cast<ushort4*>(&sKlo[srow * KSTR + d0]) = make_ushort4(kl[0], kl[1], kl[2], kl[3]);
    }
    __syncthreads();   // sync2: staged tile visible to all waves

    // ---- Issue global loads for tile t+1 now; latency hides under the WMMAs ----
    if (t0 + TT < S_LEN) {
      const float* tn = batchBase + (size_t)(t0 + TT) * D_DIM;
      #pragma unroll
      for (int i = 0; i < 8; ++i)
        reg[i] = *reinterpret_cast<const float4*>(tn + (size_t)srow * D_DIM + (squad + 4 * i) * 4);
    }

    // ---- S = Q K^T : B-operands loaded ONCE, reused by both A tiles ------------
    v8f st[2][2];   // [g][n]
    #pragma unroll
    for (int n = 0; n < 2; ++n) {
      v8f a0 = (v8f){0,0,0,0,0,0,0,0};
      v8f a1 = (v8f){0,0,0,0,0,0,0,0};
      #pragma unroll
      for (int dc = 0; dc < 4; ++dc) {
        const int kb = (n * 16 + m) * KSTR + dc * 32 + hh * 16;
        const v16bf bh = ld16(&sKhi[kb]);
        const v16bf bl = ld16(&sKlo[kb]);
        a0 = WMMA_BF16(aQhi[0][dc], bh, a0);
        a0 = WMMA_BF16(aQhi[0][dc], bl, a0);
        a0 = WMMA_BF16(aQlo[0][dc], bh, a0);
        a1 = WMMA_BF16(aQhi[1][dc], bh, a1);
        a1 = WMMA_BF16(aQhi[1][dc], bl, a1);
        a1 = WMMA_BF16(aQlo[1][dc], bh, a1);
      }
      st[0][n] = a0;
      st[1][n] = a1;
    }

    // ---- Static-max softmax: p = 2^(s*scale - M); no reductions, no rescale ----
    #pragma unroll
    for (int g = 0; g < 2; ++g) {
      #pragma unroll
      for (int r = 0; r < 8; ++r) {
        const float p0 = __builtin_amdgcn_exp2f(fmaf(st[g][0][r], scale, negM));
        const float p1 = __builtin_amdgcn_exp2f(fmaf(st[g][1][r], scale, negM));

        // stage P (D-layout -> A-layout via per-wave LDS), hi/lo packed per u32
        const int mrow = g * 16 + r + 8 * hh;
        const bf16_t h0 = (bf16_t)p0;
        const bf16_t h1 = (bf16_t)p1;
        const u16 h0b = __builtin_bit_cast(u16, h0);
        const u16 l0b = __builtin_bit_cast(u16, (bf16_t)(p0 - (float)h0));
        const u16 h1b = __builtin_bit_cast(u16, h1);
        const u16 l1b = __builtin_bit_cast(u16, (bf16_t)(p1 - (float)h1));
        sP[w][mrow * PSTRU +      m] = (u32)h0b | ((u32)l0b << 16);
        sP[w][mrow * PSTRU + 16 + m] = (u32)h1b | ((u32)l1b << 16);
      }
    }

    // Per-wave LDS RAW: DS ops of one wave execute in order (DScnt in-order);
    // only compiler reordering must be prevented — no workgroup barrier needed.
    __builtin_amdgcn_sched_barrier(0);

    // ---- O += P V and rowSum += P 1 : V B-operands loaded ONCE ----------------
    v16bf pAh[2], pAl[2];
    #pragma unroll
    for (int g = 0; g < 2; ++g) {
      const u32* pb = &sP[w][(g * 16 + m) * PSTRU];
      v8bf h0, l0, h1, l1;
      unpack8(pb + hh * 16,     h0, l0);
      unpack8(pb + hh * 16 + 8, h1, l1);
      pAh[g] = cat8(h0, h1);
      pAl[g] = cat8(l0, l1);
      // row sums on the matrix pipe: every lane of D gets the row sum
      sumAcc[g] = WMMA_BF16(pAh[g], onesB, sumAcc[g]);
      sumAcc[g] = WMMA_BF16(pAl[g], onesB, sumAcc[g]);
    }
    #pragma unroll
    for (int dt = 0; dt < 8; ++dt) {
      v16bf vh, vl;
      unpack16(&sVt[(dt * 16 + m) * VSTRU + hh * 16], vh, vl);
      o[0][dt] = WMMA_BF16(pAh[0], vh, o[0][dt]);
      o[0][dt] = WMMA_BF16(pAh[0], vl, o[0][dt]);
      o[0][dt] = WMMA_BF16(pAl[0], vh, o[0][dt]);
      o[1][dt] = WMMA_BF16(pAh[1], vh, o[1][dt]);
      o[1][dt] = WMMA_BF16(pAh[1], vl, o[1][dt]);
      o[1][dt] = WMMA_BF16(pAl[1], vh, o[1][dt]);
    }
  }

  // ---- Normalize and store (lanes 0..15 cover 16 contiguous floats per row) ----
  #pragma unroll
  for (int g = 0; g < 2; ++g) {
    float* orow = out + ((size_t)b * S_LEN + row0 + g * 16) * D_DIM;
    #pragma unroll
    for (int r = 0; r < 8; ++r) {
      const float inv  = 1.0f / sumAcc[g][r];   // row sum (same in every lane)
      const int   mrow = r + 8 * hh;
      #pragma unroll
      for (int dt = 0; dt < 8; ++dt)
        orow[(size_t)mrow * D_DIM + dt * 16 + m] = o[g][dt][r] * inv;
    }
  }
}

extern "C" void kernel_launch(void* const* d_in, const int* in_sizes, int n_in,
                              void* d_out, int out_size, void* d_ws, size_t ws_size,
                              hipStream_t stream) {
  (void)in_sizes; (void)n_in; (void)ws_size; (void)out_size;
  const float* in  = (const float*)d_in[0];
  float*       out = (float*)d_out;
  u32*         ws  = (u32*)d_ws;

  zero_ws_kernel<<<1, 1, 0, stream>>>(ws);
  maxnorm2_kernel<<<256, 128, 0, stream>>>(in, ws);   // 1024 waves x 32 rows

  dim3 grid(S_LEN / TQ, 16);   // 16 row tiles x 16 batches
  dim3 block(128);             // 4 waves of 32, 32 Q rows per wave
  flash_attn_bf16x3_kernel<<<grid, block, 0, stream>>>(in, out, ws);
}
